// GlobalPatternAware_21174188769887
// MI455X (gfx1250) — matrix operand order repaired and checked by reading, not compiled, not run on hardware
//
#include <hip/hip_runtime.h>
#include <hip/hip_bf16.h>
#include <math.h>

// ---------------------------------------------------------------------------
// Types for CDNA5 WMMA (wave32).
// ---------------------------------------------------------------------------
typedef __attribute__((ext_vector_type(16))) __bf16 v16bf;
typedef __attribute__((ext_vector_type(8)))  __bf16 bf16x8;
typedef __attribute__((ext_vector_type(8)))  float  v8f;
typedef __attribute__((ext_vector_type(4)))  float  f32x4;
typedef __attribute__((ext_vector_type(4)))  int    i32x4;

// Exact pointee type expected by the async-LDS builtins (per hipcc diag):
//   '__attribute__((__vector_size__(4 * sizeof(int)))) int __device__ *'
typedef int av4i __attribute__((vector_size(16)));
typedef __attribute__((address_space(1))) av4i* gptr_v4i;
typedef __attribute__((address_space(3))) av4i* lptr_v4i;

// Problem constants (from the reference): bs=4, w=512, n=1, nk=64, d=512.
#define DM 512          // d_model == K == N of every GEMM
#define NK 64           // number of keys
#define WW 512          // window length w
#define BS 4            // batch

#if __has_builtin(__builtin_amdgcn_global_load_async_to_lds_b128)
#define HAS_ASYNC_LDS 1
#else
#define HAS_ASYNC_LDS 0
#endif

// ---------------------------------------------------------------------------
// Kernel 1: pack a (N=512, K=512) fp32 weight (torch Linear, row = out feat)
// into bf16 WMMA B-fragment order, arranged so each k-step slice is one
// contiguous 32KB block and LDS reads are bank-conflict free:
//   layout [s(16)][nt(32)][h(2)][lane(32)][e(8)]
// B-matrix 32x16 bf16 (ISA 7.12.2): lane L holds column n = L%16; lanes 0-15
// hold K=0..7,16..23, lanes 16-31 K=8..15,24..31; fragment element E=h*8+e
// -> k = h*16 + (L/16)*8 + e of the k-step.
// ---------------------------------------------------------------------------
__global__ __launch_bounds__(256)
void pack_weight_kernel(const float* __restrict__ W, __bf16* __restrict__ out) {
  const int t  = blockIdx.x * 256 + threadIdx.x;   // 0 .. 262143
  const int e  = t & 7;
  const int L  = (t >> 3) & 31;
  const int h  = (t >> 8) & 1;
  const int nt = (t >> 9) & 31;
  const int s  = (t >> 14) & 15;
  const int n  = nt * 16 + (L & 15);
  const int k  = s * 32 + h * 16 + (L >> 4) * 8 + e;
  out[t] = (__bf16)W[n * DM + k];
}

// Cooperative async fill of one 32KB B k-slice into LDS (linear copy).
__device__ __forceinline__ void fill_slice_async(const __bf16* __restrict__ Bp,
                                                 int s, __bf16* dst, int tid) {
#if HAS_ASYNC_LDS
  const char* g = (const char*)(Bp + (size_t)s * 16384);
  char* l = (char*)dst;
#pragma unroll
  for (int j = 0; j < 4; ++j) {
    const size_t off = (size_t)(tid + j * 512) * 16;
    __builtin_amdgcn_global_load_async_to_lds_b128(
        (gptr_v4i)(unsigned long long)(g + off),
        (lptr_v4i)(unsigned int)(unsigned long long)(l + off),
        0, 0);
  }
#endif
}

// ---------------------------------------------------------------------------
// Kernel 2: C = A * W^T + bias, K = N = 512, A fp32 row-major (M x 512).
// Block: 512 threads = 16 waves, 128M x 512N tile; wave (wm,wnh) owns rows
// [wm*16,+16) x cols [wnh*256,+256) = 16 tiles of v_wmma_f32_16x16x32_bf16.
// B k-slices (32KB) are double-buffered in LDS via async global->LDS copies
// (fallback: load+ds_store) and shared by all 8 M-waves; A fragments are
// gathered from fp32 global per the ISA A layout with 1-step prefetch.
// MODE 0: fp32 row-major C[r*512+n], scaled.
// MODE 1: bf16 store transposed to [bi][w][di][m]; the 8 rows per lane are
//         8 consecutive m -> one packed 16B store.
// ---------------------------------------------------------------------------
template <int MODE>
__global__ __launch_bounds__(512)
void gemm512_kernel(const float* __restrict__ A, const __bf16* __restrict__ Bp,
                    const float* __restrict__ bias, void* __restrict__ Cout,
                    float scale) {
  __shared__ __bf16 Bs[2][16384];   // 2 x 32KB double buffer

  const int tid  = threadIdx.x;
  const int lane = tid & 31;
  const int wave = tid >> 5;
  const int wm   = wave >> 1;       // 0..7  : M-wave
  const int wnh  = wave & 1;        // 0..1  : N-half
  const int lm   = lane & 15;
  const int kh   = lane >> 4;
  const int r0   = blockIdx.x * 128 + wm * 16;

  v8f acc[16];
#pragma unroll
  for (int t = 0; t < 16; ++t) {
    const float b = bias[wnh * 256 + t * 16 + lm];
    v8f tmp = {b, b, b, b, b, b, b, b};
    acc[t] = tmp;
  }

  const float* __restrict__ arow = A + (size_t)(r0 + lm) * DM;

  // --- prologue: A frag for s=0, B slice 0 in flight -----------------------
  f32x4 a0 = *(const f32x4*)(arow + kh * 8);
  f32x4 a1 = *(const f32x4*)(arow + kh * 8 + 4);
  f32x4 a2 = *(const f32x4*)(arow + kh * 8 + 16);
  f32x4 a3 = *(const f32x4*)(arow + kh * 8 + 20);
#if HAS_ASYNC_LDS
  fill_slice_async(Bp, 0, &Bs[0][0], tid);
#else
  i32x4 creg[4];
  {
    const char* g = (const char*)Bp;
#pragma unroll
    for (int j = 0; j < 4; ++j)
      creg[j] = *(const i32x4*)(g + (size_t)(tid + j * 512) * 16);
  }
#endif

  for (int s = 0; s < 16; ++s) {
    const int d = s & 1;
#if HAS_ASYNC_LDS
    asm volatile("s_wait_asynccnt 0x0" ::: "memory");   // my slice-s copies done
#else
    {
      char* l = (char*)&Bs[d][0];
#pragma unroll
      for (int j = 0; j < 4; ++j)
        *(i32x4*)(l + (size_t)(tid + j * 512) * 16) = creg[j];
    }
#endif
    __syncthreads();   // slice s visible to all; buf[1-d] free to refill

    if (s < 15) {
#if HAS_ASYNC_LDS
      fill_slice_async(Bp, s + 1, &Bs[1 - d][0], tid);
#else
      const char* g = (const char*)(Bp + (size_t)(s + 1) * 16384);
#pragma unroll
      for (int j = 0; j < 4; ++j)
        creg[j] = *(const i32x4*)(g + (size_t)(tid + j * 512) * 16);
#endif
    }

    // A fragment for this step (16x32 bf16, ISA 7.12.2 K pattern).
    v16bf af;
#pragma unroll
    for (int j = 0; j < 4; ++j) {
      af[j]      = (__bf16)a0[j];
      af[4 + j]  = (__bf16)a1[j];
      af[8 + j]  = (__bf16)a2[j];
      af[12 + j] = (__bf16)a3[j];
    }
    // Prefetch next A fragment (overlaps the WMMA chain below).
    if (s < 15) {
      const int k0n = (s + 1) * 32 + kh * 8;
      a0 = *(const f32x4*)(arow + k0n);
      a1 = *(const f32x4*)(arow + k0n + 4);
      a2 = *(const f32x4*)(arow + k0n + 16);
      a3 = *(const f32x4*)(arow + k0n + 20);
    }

#pragma unroll
    for (int t = 0; t < 16; ++t) {
      const int nt = wnh * 16 + t;
      // Conflict-free: b128 at 16B lane stride (single base + imm offsets).
      const bf16x8 lo = *(const bf16x8*)&Bs[d][(size_t)(nt * 64 + lane) * 8];
      const bf16x8 hi = *(const bf16x8*)&Bs[d][(size_t)(nt * 64 + 32 + lane) * 8];
      v16bf bf;
#pragma unroll
      for (int j = 0; j < 8; ++j) { bf[j] = lo[j]; bf[8 + j] = hi[j]; }
      acc[t] = __builtin_amdgcn_wmma_f32_16x16x32_bf16(
          false, af, false, bf, (short)0, acc[t], false, false);
    }
  }

  // Epilogue. C/D layout: lane n = L%16, VGPR v -> row v + (L/16)*8.
  if (MODE == 0) {
    float* C = (float*)Cout;
#pragma unroll
    for (int t = 0; t < 16; ++t) {
      const int n  = wnh * 256 + t * 16 + lm;
      const int rb = r0 + kh * 8;
#pragma unroll
      for (int v = 0; v < 8; ++v)
        C[(size_t)(rb + v) * DM + n] = acc[t][v] * scale;
    }
  } else {
    __bf16* C = (__bf16*)Cout;
#pragma unroll
    for (int t = 0; t < 16; ++t) {
      const int n  = wnh * 256 + t * 16 + lm;   // di
      const int r  = r0 + kh * 8;               // 8 consecutive rows = 8 m
      const int m  = r & 63;
      const int w  = (r >> 6) & 511;
      const int bi = r >> 15;
      bf16x8 o;
#pragma unroll
      for (int v = 0; v < 8; ++v) o[v] = (__bf16)(acc[t][v] * scale);
      *(bf16x8*)(C + ((size_t)((bi * WW + w) * DM + n)) * NK + m) = o;
    }
  }
}

// ---------------------------------------------------------------------------
// Kernel 3: per-(bi,di) attention over nk=64 keys, contraction over w=512.
// Kp/Vp layout [bi][w][di][m] bf16 -> lane(di) reads 8 keys per b128 load.
// Block: 256 threads = 8 waves; wave = w-slice (wp), lane = di (32-wide).
// LDS [wp][m][dl] partials (conflict-free), reduce + softmax, V pass writes
// X[w][bi][di] (the row layout the final GEMM consumes).
// ---------------------------------------------------------------------------
__global__ __launch_bounds__(256)
void attn_kernel(const float* __restrict__ Qp, const __bf16* __restrict__ Kp,
                 const __bf16* __restrict__ Vp, float* __restrict__ X) {
  __shared__ float part[8][64][32];   // 64 KB
  const int bi = blockIdx.x >> 4;
  const int dt = blockIdx.x & 15;
  const int t  = threadIdx.x;
  const int dl = t & 31;
  const int wp = t >> 5;
  const int di = dt * 32 + dl;

  float s[NK];
#pragma unroll
  for (int m = 0; m < NK; ++m) s[m] = 0.f;

  const int w0 = wp * 64;
  for (int w = w0; w < w0 + 64; ++w) {
    const float q = Qp[(size_t)(bi * WW + w) * DM + di];   // already /sqrt(d)
    const __bf16* kp = Kp + ((size_t)((bi * WW + w) * DM + di)) * NK;
#pragma unroll
    for (int mg = 0; mg < 8; ++mg) {
      const bf16x8 kv = *(const bf16x8*)(kp + mg * 8);
#pragma unroll
      for (int j = 0; j < 8; ++j) s[mg * 8 + j] += q * (float)kv[j];
    }
  }
#pragma unroll
  for (int m = 0; m < NK; ++m) part[wp][m][dl] = s[m];
  __syncthreads();

  if (wp == 0) {   // softmax per di over the 64 keys
    float r[NK];
    float mx = -3.4e38f;
#pragma unroll
    for (int m = 0; m < NK; ++m) {
      float v = 0.f;
#pragma unroll
      for (int p = 0; p < 8; ++p) v += part[p][m][dl];
      r[m] = v;
      mx = fmaxf(mx, v);
    }
    float sum = 0.f;
#pragma unroll
    for (int m = 0; m < NK; ++m) { r[m] = __expf(r[m] - mx); sum += r[m]; }
    const float inv = 1.f / sum;
#pragma unroll
    for (int m = 0; m < NK; ++m) part[0][m][dl] = r[m] * inv;
  }
  __syncthreads();

  float a[NK];
#pragma unroll
  for (int m = 0; m < NK; ++m) a[m] = part[0][m][dl];

  for (int w = w0; w < w0 + 64; ++w) {
    const __bf16* vp = Vp + ((size_t)((bi * WW + w) * DM + di)) * NK;
    float o = 0.f;
#pragma unroll
    for (int mg = 0; mg < 8; ++mg) {
      const bf16x8 vv = *(const bf16x8*)(vp + mg * 8);
#pragma unroll
      for (int j = 0; j < 8; ++j) o += a[mg * 8 + j] * (float)vv[j];
    }
    X[(size_t)(w * BS + bi) * DM + di] = o;   // X[w][bi][di]
  }
}

// ---------------------------------------------------------------------------
// Launcher. Workspace: Qproj(4MB) | Kp bf16(128MB) | Vp bf16(128MB) |
// X(4MB) | 4 packed weights (512KB each)  -> ~266 MiB total.
// ---------------------------------------------------------------------------
extern "C" void kernel_launch(void* const* d_in, const int* in_sizes, int n_in,
                              void* d_out, int out_size, void* d_ws, size_t ws_size,
                              hipStream_t stream) {
  (void)in_sizes; (void)n_in; (void)out_size; (void)ws_size;
  const float* Q  = (const float*)d_in[0];
  const float* K  = (const float*)d_in[1];
  const float* V  = (const float*)d_in[2];
  const float* Wq = (const float*)d_in[3];
  const float* bq = (const float*)d_in[4];
  const float* Wk = (const float*)d_in[5];
  const float* bk = (const float*)d_in[6];
  const float* Wv = (const float*)d_in[7];
  const float* bv = (const float*)d_in[8];
  const float* Wp = (const float*)d_in[9];
  const float* bp = (const float*)d_in[10];

  char* ws = (char*)d_ws;
  const size_t SZ_PROJ = (size_t)2048 * DM * sizeof(float);     // 4 MB
  const size_t SZ_KV   = (size_t)BS * WW * DM * NK * 2;         // 128 MB
  const size_t SZ_PK   = (size_t)DM * DM * 2;                   // 512 KB
  float*  Qproj = (float*)(ws);
  __bf16* Kp    = (__bf16*)(ws + SZ_PROJ);
  __bf16* Vp    = (__bf16*)(ws + SZ_PROJ + SZ_KV);
  float*  X     = (float*)(ws + SZ_PROJ + 2 * SZ_KV);
  __bf16* BpQ   = (__bf16*)(ws + 2 * SZ_PROJ + 2 * SZ_KV);
  __bf16* BpK   = (__bf16*)((char*)BpQ + SZ_PK);
  __bf16* BpV   = (__bf16*)((char*)BpK + SZ_PK);
  __bf16* BpP   = (__bf16*)((char*)BpV + SZ_PK);

  // 1) pack weights into k-slice-contiguous WMMA B-fragment order (bf16)
  pack_weight_kernel<<<1024, 256, 0, stream>>>(Wq, BpQ);
  pack_weight_kernel<<<1024, 256, 0, stream>>>(Wk, BpK);
  pack_weight_kernel<<<1024, 256, 0, stream>>>(Wv, BpV);
  pack_weight_kernel<<<1024, 256, 0, stream>>>(Wp, BpP);

  // 2) projections (bias folded into acc init; 1/sqrt(512) folded into Q)
  const float qscale = 0.044194173824159216f;                   // 1/sqrt(512)
  gemm512_kernel<0><<<16,   512, 0, stream>>>(Q, BpQ, bq, (void*)Qproj, qscale);
  gemm512_kernel<1><<<1024, 512, 0, stream>>>(K, BpK, bk, (void*)Kp, 1.0f);
  gemm512_kernel<1><<<1024, 512, 0, stream>>>(V, BpV, bv, (void*)Vp, 1.0f);

  // 3) attention over the 64 keys, contraction over w
  attn_kernel<<<64, 256, 0, stream>>>(Qproj, Kp, Vp, X);

  // 4) output projection -> d_out (w, bs, d) fp32 row-major
  gemm512_kernel<0><<<16, 512, 0, stream>>>(X, BpP, bp, d_out, 1.0f);
}
